// SE3GPCRGNN_18330920419495
// MI455X (gfx1250) — compile-verified, matrix operand order checked by reading, not run on hardware
//
#include <hip/hip_runtime.h>
#include <hip/hip_bf16.h>
#include <math.h>

typedef __attribute__((ext_vector_type(16))) _Float16 v16h;
typedef __attribute__((ext_vector_type(8)))  float    v8f;

#define N_NODES  50000
#define N_EDGES  1000000
#define NODE_IN  1281
#define HDIM     256
#define NLAYERS  4
#define NRADIAL  16
#define MSG_IN   275
#define KCH_NE   41     /* ceil(1281/32) -> Kpad 1312 */
#define KCH_MSG  9      /* 288/32 */
#define KCH_H    8      /* 256/32 */
#define CHUNK_ELEMS 8192 /* 16 ntiles * 32 lanes * 16 halves per K-chunk of packed W */

#if defined(__has_builtin)
#if __has_builtin(__builtin_amdgcn_sched_barrier)
#define SCHED_BARRIER() __builtin_amdgcn_sched_barrier(0)
#endif
#endif
#ifndef SCHED_BARRIER
#define SCHED_BARRIER()
#endif

__device__ __forceinline__ float silu_f(float x) { return x / (1.0f + __expf(-x)); }

__device__ __forceinline__ void atomic_add_f(float* p, float v) {
  __hip_atomic_fetch_add(p, v, __ATOMIC_RELAXED, __HIP_MEMORY_SCOPE_AGENT);
}

// Async gather of 16 bytes from global memory directly into LDS (per-lane
// addresses; tracked by ASYNCcnt). CDNA5 GLOBAL_LOAD_ASYNC_TO_LDS_B128.
__device__ __forceinline__ void async_gather_b128(unsigned lds_off, const void* gptr) {
  unsigned long long ga = (unsigned long long)(uintptr_t)gptr;
  asm volatile("global_load_async_to_lds_b128 %0, %1, off"
               :: "v"(lds_off), "v"(ga)
               : "memory");
}
__device__ __forceinline__ void wait_asynccnt0() {
  asm volatile("s_wait_asynccnt 0x0" ::: "memory");
}

// ---------------------------------------------------------------------------
// Pack a (K x 256) f32 weight matrix into f16 WMMA B-fragments.
// Fragment order: [kchunk c][ntile t][lane l][half h], where within a 32-K
// chunk, lane group g = l>>4 and half h map to k_local per the CDNA5 16-bit
// operand layout: h<8 -> k=8g+h ; h>=8 -> k=16+8g+(h-8). lane&15 = column.
// ---------------------------------------------------------------------------
__global__ __launch_bounds__(256) void pack_w_kernel(const float* __restrict__ W,
                                                     _Float16* __restrict__ dst,
                                                     int K, int nchunks) {
  int idx = blockIdx.x * blockDim.x + threadIdx.x;
  if (idx >= (nchunks << 13)) return;
  int hh = idx & 15;
  int l  = (idx >> 4) & 31;
  int tl = (idx >> 9) & 15;
  int c  = idx >> 13;
  int g  = l >> 4;
  int kl = (hh < 8) ? (8 * g + hh) : (16 + 8 * g + (hh - 8));
  int k  = c * 32 + kl;
  int n  = tl * 16 + (l & 15);
  float v = (k < K) ? W[(size_t)k * HDIM + n] : 0.0f;
  dst[idx] = (_Float16)v;
}

// ---------------------------------------------------------------------------
// Edge geometry: sh (3) + radial RBF (16) as f16, padded to 32 per edge.
// ---------------------------------------------------------------------------
__global__ __launch_bounds__(256) void edge_geom_kernel(const float* __restrict__ pos,
                                                        const int* __restrict__ src,
                                                        const int* __restrict__ dst,
                                                        _Float16* __restrict__ efeat) {
  int e = blockIdx.x * blockDim.x + threadIdx.x;
  if (e >= N_EDGES) return;
  int s = src[e], d = dst[e];
  float rx = pos[3 * d + 0] - pos[3 * s + 0];
  float ry = pos[3 * d + 1] - pos[3 * s + 1];
  float rz = pos[3 * d + 2] - pos[3 * s + 2];
  float dist = sqrtf(rx * rx + ry * ry + rz * rz);
  float inv = 1.0f / fmaxf(dist, 1e-12f);
  _Float16* o = efeat + (size_t)e * 32;
  o[0] = (_Float16)(rx * inv);
  o[1] = (_Float16)(ry * inv);
  o[2] = (_Float16)(rz * inv);
  const float width = 0.1953125f;  // 0.5*(10/16)^2
#pragma unroll
  for (int j = 0; j < NRADIAL; ++j) {
    float cc = (10.0f / 15.0f) * (float)j;
    float dd = dist - cc;
    o[3 + j] = (_Float16)__expf(-width * dd * dd);
  }
#pragma unroll
  for (int j = 19; j < 32; ++j) o[j] = (_Float16)0.0f;
}

__global__ __launch_bounds__(256) void zero_kernel(float* __restrict__ p, int n) {
  int i = blockIdx.x * blockDim.x + threadIdx.x;
  if (i < n) p[i] = 0.0f;
}

// ---------------------------------------------------------------------------
// Stage one 64x32 f32 tile -> f16 LDS (stride 40 halves). Uniform branch on
// full vs tail chunk: full chunks have NO per-element predication; K%8==0
// additionally uses float4 vector loads.
// ---------------------------------------------------------------------------
__device__ __forceinline__ void stage_chunk(const float* __restrict__ A,
                                            _Float16* __restrict__ sbuf,
                                            int r0, int M, int K, int c, int t) {
  int row = t >> 2;
  int cg = (t & 3) * 8;
  int grow = min(r0 + row, M - 1);
  const float* ap = A + (size_t)grow * K + c * 32 + cg;
  _Float16 tmp[8];
  if (c * 32 + 32 <= K) {          // full chunk (uniform condition)
    if ((K & 7) == 0) {            // rows 32B-aligned -> vector loads
      float4 v0 = ((const float4*)ap)[0];
      float4 v1 = ((const float4*)ap)[1];
      tmp[0] = (_Float16)v0.x; tmp[1] = (_Float16)v0.y;
      tmp[2] = (_Float16)v0.z; tmp[3] = (_Float16)v0.w;
      tmp[4] = (_Float16)v1.x; tmp[5] = (_Float16)v1.y;
      tmp[6] = (_Float16)v1.z; tmp[7] = (_Float16)v1.w;
    } else {
#pragma unroll
      for (int i = 0; i < 8; ++i) tmp[i] = (_Float16)ap[i];
    }
  } else {                          // tail chunk only
#pragma unroll
    for (int i = 0; i < 8; ++i) {
      int k = c * 32 + cg + i;
      tmp[i] = (k < K) ? (_Float16)ap[i] : (_Float16)0.0f;
    }
  }
  *(float4*)(sbuf + row * 40 + cg) = *(float4*)tmp;
}

// ---------------------------------------------------------------------------
// Dense GEMM: out[M x 256] = epilogue(A[M x K](f32) @ Wp + bias)
// 64 rows/block, 8 waves; wave = (mtile 0..3, nhalf 0..1) -> 16x128 tile.
// Double-buffered LDS staging (1 barrier/chunk). B-fragment load batch is
// fenced from the WMMA batch with a sched_barrier so all 16 b128 loads are
// in flight before the first WMMA consumes them.
// ---------------------------------------------------------------------------
__global__ __launch_bounds__(256) void dense_gemm_kernel(const float* __restrict__ A,
                                                         const _Float16* __restrict__ Wp,
                                                         const float* __restrict__ bias,
                                                         float* __restrict__ out,
                                                         int M, int K, int nchunks,
                                                         int do_silu) {
  __shared__ _Float16 sA[2][64 * 40];
  const int t = threadIdx.x;
  const int r0 = blockIdx.x * 64;
  const int wave = t >> 5, lane = t & 31;
  const int mtile = wave >> 1, nhalf = wave & 1;
  const int g = lane >> 4, lm = lane & 15;

  v8f acc[8];
#pragma unroll
  for (int i = 0; i < 8; ++i) acc[i] = (v8f){0.f, 0.f, 0.f, 0.f, 0.f, 0.f, 0.f, 0.f};

  stage_chunk(A, sA[0], r0, M, K, 0, t);

  for (int c = 0; c < nchunks; ++c) {
    __syncthreads();
    const _Float16* cbuf = sA[c & 1];
    if (c + 1 < nchunks) stage_chunk(A, sA[(c + 1) & 1], r0, M, K, c + 1, t);

    v16h a;
    {
      const _Float16* arow = cbuf + (mtile * 16 + lm) * 40;
      ((float4*)&a)[0] = *(const float4*)(arow + 8 * g);
      ((float4*)&a)[1] = *(const float4*)(arow + 16 + 8 * g);
    }
    v16h b[8];
#pragma unroll
    for (int tt = 0; tt < 8; ++tt) {
      const _Float16* bp = Wp + (((size_t)c * 16 + (nhalf * 8 + tt)) * 32 + lane) * 16;
      ((float4*)&b[tt])[0] = *(const float4*)(bp);
      ((float4*)&b[tt])[1] = *(const float4*)(bp + 8);
    }
    SCHED_BARRIER();
#pragma unroll
    for (int tt = 0; tt < 8; ++tt)
      acc[tt] = __builtin_amdgcn_wmma_f32_16x16x32_f16(false, a, false, b[tt],
                                                       (short)0, acc[tt], false, false);
    SCHED_BARRIER();
  }
  // epilogue: C layout -> lane lm = col, row m = mtile*16 + 8*g + r
#pragma unroll
  for (int tt = 0; tt < 8; ++tt) {
    int ncol = (nhalf * 8 + tt) * 16 + lm;
    float bv = bias[ncol];
#pragma unroll
    for (int r = 0; r < 8; ++r) {
      int row = r0 + mtile * 16 + 8 * g + r;
      if (row < M) {
        float v = acc[tt][r] + bv;
        if (do_silu) v = silu_f(v);
        out[(size_t)row * HDIM + ncol] = v;
      }
    }
  }
}

// ---------------------------------------------------------------------------
// Edge-message GEMM + scatter-add:
//  agg[dst[e]] += concat(h16[src[e]], sh, radial) @ convWp + bias
// 64 edges/block; 64x288 f16 A tile (stride 296 halves). h rows are gathered
// straight from the persistent f16 mirror into LDS with CDNA5 async
// global->LDS b128 copies (no VGPR round-trip, no conversion), then 9 K-chunks
// of 8 back-to-back WMMAs per wave.
// ---------------------------------------------------------------------------
__global__ __launch_bounds__(256) void edge_msg_kernel(const _Float16* __restrict__ h16,
                                                       const _Float16* __restrict__ efeat,
                                                       const int* __restrict__ src,
                                                       const int* __restrict__ dst,
                                                       const _Float16* __restrict__ Wp,
                                                       const float* __restrict__ bias,
                                                       float* __restrict__ agg, int nE) {
  __shared__ _Float16 sA[64 * 296];
  const int t = threadIdx.x;
  const int e0 = blockIdx.x * 64;

  // async-gather h16[src] rows: thread (el = t&63, q = t>>6) owns 64 halves
  {
    int el = t & 63;
    int q = t >> 6;
    int e = min(e0 + el, nE - 1);
    int srow = src[e];
    const _Float16* hp = h16 + (size_t)srow * HDIM + q * 64;
    unsigned lds_off = (unsigned)(uintptr_t)(sA + el * 296 + q * 64);
#pragma unroll
    for (int i = 0; i < 8; ++i)
      async_gather_b128(lds_off + i * 16, hp + i * 8);
  }
  // append edge features at k = 256..287 (19 valid + 13 zero pad, prepacked)
  if (t < 64) {
    int e = min(e0 + t, nE - 1);
    const float4* fp = (const float4*)(efeat + (size_t)e * 32);
    float4* o = (float4*)(sA + t * 296 + 256);
#pragma unroll
    for (int i = 0; i < 4; ++i) o[i] = fp[i];
  }
  wait_asynccnt0();
  __syncthreads();

  const int wave = t >> 5, lane = t & 31;
  const int mtile = wave >> 1, nhalf = wave & 1;
  const int g = lane >> 4, lm = lane & 15;

  v8f acc[8];
#pragma unroll
  for (int i = 0; i < 8; ++i) acc[i] = (v8f){0.f, 0.f, 0.f, 0.f, 0.f, 0.f, 0.f, 0.f};

#pragma unroll
  for (int c = 0; c < KCH_MSG; ++c) {
    v16h a;
    const _Float16* arow = sA + (mtile * 16 + lm) * 296 + c * 32;
    ((float4*)&a)[0] = *(const float4*)(arow + 8 * g);
    ((float4*)&a)[1] = *(const float4*)(arow + 16 + 8 * g);
    v16h b[8];
#pragma unroll
    for (int tt = 0; tt < 8; ++tt) {
      const _Float16* bp = Wp + (((size_t)c * 16 + (nhalf * 8 + tt)) * 32 + lane) * 16;
      ((float4*)&b[tt])[0] = *(const float4*)(bp);
      ((float4*)&b[tt])[1] = *(const float4*)(bp + 8);
    }
    SCHED_BARRIER();
#pragma unroll
    for (int tt = 0; tt < 8; ++tt)
      acc[tt] = __builtin_amdgcn_wmma_f32_16x16x32_f16(false, a, false, b[tt],
                                                       (short)0, acc[tt], false, false);
    SCHED_BARRIER();
  }

  // scatter-add messages (bias included per-edge, as in the reference)
#pragma unroll
  for (int tt = 0; tt < 8; ++tt) {
    int ncol = (nhalf * 8 + tt) * 16 + lm;
    float bv = bias[ncol];
#pragma unroll
    for (int r = 0; r < 8; ++r) {
      int e = e0 + mtile * 16 + 8 * g + r;
      if (e < nE) {
        int drow = dst[e];
        atomic_add_f(&agg[(size_t)drow * HDIM + ncol], acc[tt][r] + bv);
      }
    }
  }
}

// ---------------------------------------------------------------------------
// LayerNorm over H=256, one block per row; also emits the f16 mirror.
// ---------------------------------------------------------------------------
__global__ __launch_bounds__(256) void ln_kernel(float* __restrict__ x,
                                                 const float* __restrict__ gam,
                                                 const float* __restrict__ bet,
                                                 _Float16* __restrict__ h16) {
  __shared__ float s1[256], s2[256];
  int n = blockIdx.x, t = threadIdx.x;
  float v = x[(size_t)n * HDIM + t];
  s1[t] = v; s2[t] = v * v;
  __syncthreads();
  for (int s = 128; s > 0; s >>= 1) {
    if (t < s) { s1[t] += s1[t + s]; s2[t] += s2[t + s]; }
    __syncthreads();
  }
  float mean = s1[0] / 256.0f;
  float var = s2[0] / 256.0f - mean * mean;
  float r = (v - mean) * rsqrtf(var + 1e-5f) * gam[t] + bet[t];
  x[(size_t)n * HDIM + t] = r;
  h16[(size_t)n * HDIM + t] = (_Float16)r;
}

// h = LN2( h + LN1(silu(agg)) ); also refreshes the f16 mirror
__global__ __launch_bounds__(256) void node_update_kernel(float* __restrict__ h,
                                                          _Float16* __restrict__ h16,
                                                          const float* __restrict__ agg,
                                                          const float* __restrict__ cg,
                                                          const float* __restrict__ cb,
                                                          const float* __restrict__ lg,
                                                          const float* __restrict__ lb) {
  __shared__ float s1[256], s2[256];
  int n = blockIdx.x, t = threadIdx.x;
  float v = silu_f(agg[(size_t)n * HDIM + t]);
  s1[t] = v; s2[t] = v * v;
  __syncthreads();
  for (int s = 128; s > 0; s >>= 1) {
    if (t < s) { s1[t] += s1[t + s]; s2[t] += s2[t + s]; }
    __syncthreads();
  }
  float mean = s1[0] / 256.0f;
  float var = s2[0] / 256.0f - mean * mean;
  __syncthreads();
  float hn = (v - mean) * rsqrtf(var + 1e-5f) * cg[t] + cb[t];
  float r = h[(size_t)n * HDIM + t] + hn;
  s1[t] = r; s2[t] = r * r;
  __syncthreads();
  for (int s = 128; s > 0; s >>= 1) {
    if (t < s) { s1[t] += s1[t + s]; s2[t] += s2[t + s]; }
    __syncthreads();
  }
  mean = s1[0] / 256.0f;
  var = s2[0] / 256.0f - mean * mean;
  float o = (r - mean) * rsqrtf(var + 1e-5f) * lg[t] + lb[t];
  h[(size_t)n * HDIM + t] = o;
  h16[(size_t)n * HDIM + t] = (_Float16)o;
}

// attention gate scalar per node: silu(h @ pgW1 + b1) @ pgW2 + b2
__global__ __launch_bounds__(64) void gate_kernel(const float* __restrict__ h,
                                                  const float* __restrict__ w1,
                                                  const float* __restrict__ b1,
                                                  const float* __restrict__ w2,
                                                  const float* __restrict__ b2,
                                                  float* __restrict__ gates) {
  __shared__ float red[64];
  int n = blockIdx.x, j = threadIdx.x;
  const float* hr = h + (size_t)n * HDIM;
  float a = b1[j];
  for (int k = 0; k < HDIM; ++k) a += hr[k] * w1[k * 64 + j];
  red[j] = silu_f(a) * w2[j];
  __syncthreads();
  for (int s = 32; s > 0; s >>= 1) {
    if (j < s) red[j] += red[j + s];
    __syncthreads();
  }
  if (j == 0) gates[n] = red[0] + b2[0];
}

// softmax stats over all nodes; also zero graph-emb accumulator
__global__ __launch_bounds__(256) void gate_reduce_kernel(const float* __restrict__ gates,
                                                          float* __restrict__ scal,
                                                          float* __restrict__ gemb) {
  __shared__ float red[256];
  int t = threadIdx.x;
  float m = -1e30f;
  for (int i = t; i < N_NODES; i += 256) m = fmaxf(m, gates[i]);
  red[t] = m;
  __syncthreads();
  for (int s = 128; s > 0; s >>= 1) {
    if (t < s) red[t] = fmaxf(red[t], red[t + s]);
    __syncthreads();
  }
  float gmax = red[0];
  __syncthreads();
  float sum = 0.0f;
  for (int i = t; i < N_NODES; i += 256) sum += __expf(gates[i] - gmax);
  red[t] = sum;
  __syncthreads();
  for (int s = 128; s > 0; s >>= 1) {
    if (t < s) red[t] += red[t + s];
    __syncthreads();
  }
  if (t == 0) { scal[0] = gmax; scal[1] = red[0]; }
  gemb[t] = 0.0f;
}

__global__ __launch_bounds__(256) void pool_kernel(const float* __restrict__ h,
                                                   const float* __restrict__ gates,
                                                   const float* __restrict__ scal,
                                                   float* __restrict__ gemb) {
  int t = threadIdx.x;
  float gmax = scal[0], inv = 1.0f / scal[1];
  float acc = 0.0f;
  for (int n = blockIdx.x; n < N_NODES; n += gridDim.x)
    acc += h[(size_t)n * HDIM + t] * __expf(gates[n] - gmax);
  atomic_add_f(&gemb[t], acc * inv);
}

__global__ __launch_bounds__(256) void heads_kernel(const float* __restrict__ gemb,
                                                    const float* __restrict__ clW1,
                                                    const float* __restrict__ clb1,
                                                    const float* __restrict__ clW2,
                                                    const float* __restrict__ clb2,
                                                    const float* __restrict__ prW1,
                                                    const float* __restrict__ prb1,
                                                    const float* __restrict__ prW2,
                                                    const float* __restrict__ prb2,
                                                    float* __restrict__ out) {
  __shared__ float ge[256], t1[256], red[128];
  int t = threadIdx.x;
  ge[t] = gemb[t];
  __syncthreads();
  // classifier
  if (t < 128) {
    float a = clb1[t];
    for (int k = 0; k < 256; ++k) a += ge[k] * clW1[k * 128 + t];
    t1[t] = silu_f(a);
  }
  __syncthreads();
  if (t < 4) {
    float a = clb2[t];
    for (int k = 0; k < 128; ++k) a += t1[k] * clW2[k * 4 + t];
    out[t] = a;
  }
  __syncthreads();
  // projector
  float a = prb1[t];
  for (int k = 0; k < 256; ++k) a += ge[k] * prW1[k * 256 + t];
  t1[t] = silu_f(a);
  __syncthreads();
  float pv = 0.0f;
  if (t < 128) {
    pv = prb2[t];
    for (int k = 0; k < 256; ++k) pv += t1[k] * prW2[k * 128 + t];
    red[t] = pv * pv;
  }
  __syncthreads();
  for (int s = 64; s > 0; s >>= 1) {
    if (t < s) red[t] += red[t + s];
    __syncthreads();
  }
  float nrm = fmaxf(sqrtf(red[0]), 1e-12f);
  if (t < 128) out[4 + t] = pv / nrm;
  out[132 + t] = ge[t];
}

// ---------------------------------------------------------------------------
extern "C" void kernel_launch(void* const* d_in, const int* in_sizes, int n_in,
                              void* d_out, int out_size, void* d_ws, size_t ws_size,
                              hipStream_t stream) {
  (void)in_sizes; (void)n_in; (void)out_size; (void)ws_size;
  const float* x      = (const float*)d_in[0];
  const float* pos    = (const float*)d_in[1];
  const int*   eidx   = (const int*)d_in[2];
  const int*   esrc   = eidx;
  const int*   edst   = eidx + N_EDGES;
  const float* ne_W1  = (const float*)d_in[3];
  const float* ne_b1  = (const float*)d_in[4];
  const float* ne_W2  = (const float*)d_in[5];
  const float* ne_b2  = (const float*)d_in[6];
  const float* ne_g   = (const float*)d_in[7];
  const float* ne_bt  = (const float*)d_in[8];
  const float* convW  = (const float*)d_in[9];
  const float* convB  = (const float*)d_in[10];
  const float* convG  = (const float*)d_in[11];
  const float* convBt = (const float*)d_in[12];
  const float* lnG    = (const float*)d_in[13];
  const float* lnB    = (const float*)d_in[14];
  const float* pgW1   = (const float*)d_in[15];
  const float* pgb1   = (const float*)d_in[16];
  const float* pgW2   = (const float*)d_in[17];
  const float* pgb2   = (const float*)d_in[18];
  const float* clW1   = (const float*)d_in[19];
  const float* clb1   = (const float*)d_in[20];
  const float* clW2   = (const float*)d_in[21];
  const float* clb2   = (const float*)d_in[22];
  const float* prW1   = (const float*)d_in[23];
  const float* prb1   = (const float*)d_in[24];
  const float* prW2   = (const float*)d_in[25];
  const float* prb2   = (const float*)d_in[26];
  float* out = (float*)d_out;

  char* ws = (char*)d_ws;
  size_t off = 0;
  auto wsa = [&](size_t bytes) {
    void* p = ws + off;
    off += (bytes + 255) & ~(size_t)255;
    return p;
  };
  float*    h     = (float*)wsa((size_t)N_NODES * HDIM * 4);
  _Float16* h16   = (_Float16*)wsa((size_t)N_NODES * HDIM * 2);
  float*    agg   = (float*)wsa((size_t)N_NODES * HDIM * 4);  // also encoder mid
  _Float16* efeat = (_Float16*)wsa((size_t)N_EDGES * 32 * 2);
  _Float16* w1p   = (_Float16*)wsa((size_t)KCH_NE * CHUNK_ELEMS * 2);
  _Float16* w2p   = (_Float16*)wsa((size_t)KCH_H * CHUNK_ELEMS * 2);
  _Float16* cwp   = (_Float16*)wsa((size_t)NLAYERS * KCH_MSG * CHUNK_ELEMS * 2);
  float*    gates = (float*)wsa((size_t)N_NODES * 4);
  float*    scal  = (float*)wsa(64);
  float*    gemb  = (float*)wsa(1024);

  // 1) repack weights to WMMA fragment layout (f16)
  pack_w_kernel<<<(KCH_NE * CHUNK_ELEMS + 255) / 256, 256, 0, stream>>>(ne_W1, w1p, NODE_IN, KCH_NE);
  pack_w_kernel<<<(KCH_H * CHUNK_ELEMS + 255) / 256, 256, 0, stream>>>(ne_W2, w2p, HDIM, KCH_H);
  for (int l = 0; l < NLAYERS; ++l)
    pack_w_kernel<<<(KCH_MSG * CHUNK_ELEMS + 255) / 256, 256, 0, stream>>>(
        convW + (size_t)l * MSG_IN * HDIM, cwp + (size_t)l * KCH_MSG * CHUNK_ELEMS, MSG_IN, KCH_MSG);

  // 2) edge geometry features
  edge_geom_kernel<<<(N_EDGES + 255) / 256, 256, 0, stream>>>(pos, esrc, edst, efeat);

  // 3) node encoder: silu(x@W1+b1)@W2+b2, LN (LN also emits the f16 mirror)
  dense_gemm_kernel<<<(N_NODES + 63) / 64, 256, 0, stream>>>(x, w1p, ne_b1, agg,
                                                             N_NODES, NODE_IN, KCH_NE, 1);
  dense_gemm_kernel<<<(N_NODES + 63) / 64, 256, 0, stream>>>(agg, w2p, ne_b2, h,
                                                             N_NODES, HDIM, KCH_H, 0);
  ln_kernel<<<N_NODES, 256, 0, stream>>>(h, ne_g, ne_bt, h16);

  // 4) message-passing layers
  for (int l = 0; l < NLAYERS; ++l) {
    zero_kernel<<<(N_NODES * HDIM + 255) / 256, 256, 0, stream>>>(agg, N_NODES * HDIM);
    edge_msg_kernel<<<(N_EDGES + 63) / 64, 256, 0, stream>>>(
        h16, efeat, esrc, edst, cwp + (size_t)l * KCH_MSG * CHUNK_ELEMS,
        convB + (size_t)l * HDIM, agg, N_EDGES);
    node_update_kernel<<<N_NODES, 256, 0, stream>>>(h, h16, agg,
        convG + (size_t)l * HDIM, convBt + (size_t)l * HDIM,
        lnG + (size_t)l * HDIM, lnB + (size_t)l * HDIM);
  }

  // 5) attention pooling
  gate_kernel<<<N_NODES, 64, 0, stream>>>(h, pgW1, pgb1, pgW2, pgb2, gates);
  gate_reduce_kernel<<<1, 256, 0, stream>>>(gates, scal, gemb);
  pool_kernel<<<256, 256, 0, stream>>>(h, gates, scal, gemb);

  // 6) heads -> d_out = [logits(4) | proj(128) | graph_emb(256)]
  heads_kernel<<<1, 256, 0, stream>>>(gemb, clW1, clb1, clW2, clb2,
                                      prW1, prb1, prW2, prb2, out);
}
